// LinearTransformerBlock_27041114096177
// MI455X (gfx1250) — matrix operand order repaired and checked by reading, not compiled, not run on hardware
//
#include <hip/hip_runtime.h>

typedef __bf16 bf16;
typedef __attribute__((ext_vector_type(16))) __bf16 v16bf;
typedef __attribute__((ext_vector_type(8)))  float  v8f;
typedef unsigned int uint4v __attribute__((ext_vector_type(4)));
typedef int          int8v  __attribute__((ext_vector_type(8)));
typedef int          int4v  __attribute__((ext_vector_type(4)));

#define WMMA_BF16(a, b, c) \
  __builtin_amdgcn_wmma_f32_16x16x32_bf16(false, (a), false, (b), (short)0, (c), false, false)

constexpr int Bc = 2, S = 4096, C = 2048, H = 16, D = 128;
constexpr int Mtot = Bc * S;          // 8192
constexpr int LDA  = 40;              // bf16 units; 80B row stride, 16B-aligned chunks

#if __has_builtin(__builtin_amdgcn_tensor_load_to_lds)
#define HAVE_TDM 1
#else
#define HAVE_TDM 0
#endif

union Frag { uint4 u[2]; v16bf v; };

__device__ __forceinline__ v16bf ld_frag(const bf16* p0, const bf16* p1) {
  Frag f;
  f.u[0] = *reinterpret_cast<const uint4*>(p0);
  f.u[1] = *reinterpret_cast<const uint4*>(p1);
  return f.v;
}

#if HAVE_TDM
// TDM: DMA a 32(dim0, contiguous) x 128(dim1) bf16 tile, row stride C elements,
// into LDS with pad_interval=16 DWORDs / pad_amount=4 DWORDs -> LDS row stride
// of 40 bf16, exactly matching the LDA layout the WMMA fragment loads expect.
__device__ __forceinline__ void tdm_load_tile(const bf16* gptr, unsigned lds_off,
                                              int tensor_rows) {
  const unsigned long long ga = (unsigned long long)(uintptr_t)gptr;
  uint4v g0;
  g0.x = 1u;                                        // count = 1 (valid descriptor)
  g0.y = lds_off;                                   // lds_addr
  g0.z = (unsigned)(ga & 0xffffffffu);              // global_addr[31:0]
  g0.w = (unsigned)((ga >> 32) & 0x01ffffffu) | 0x80000000u;  // addr[56:32] | type=2

  int8v g1;
  g1[0] = (1 << 16)            // data_size = 1 (2 bytes)
        | (1 << 20)            // pad_enable
        | (3 << 22)            // pad_interval: 16 DWORDs (one 32-elem bf16 row)
        | (3 << 25);           // pad_amount:   4 DWORDs (8 bf16 of padding)
  g1[1] = (int)((unsigned)(C & 0xffff) << 16);                    // tensor_dim0 lo16
  g1[2] = (int)(((C >> 16) & 0xffff) | ((tensor_rows & 0xffff) << 16));
  g1[3] = (int)(((tensor_rows >> 16) & 0xffff) | (32u << 16));    // tile_dim0 = 32
  g1[4] = 128;                                                    // tile_dim1 = 128
  g1[5] = C;                                                      // tensor_dim0_stride
  g1[6] = 0;
  g1[7] = 0;

  int4v z4 = {0, 0, 0, 0};
#if defined(__clang_major__) && (__clang_major__ >= 23)
  int8v z8 = {0, 0, 0, 0, 0, 0, 0, 0};
  __builtin_amdgcn_tensor_load_to_lds(g0, g1, z4, z4, z8, 0);
#else
  __builtin_amdgcn_tensor_load_to_lds(g0, g1, z4, z4, 0);
#endif
}
#endif

// ---------------------------------------------------------------------------
// fp32 -> bf16 pack (one-time pre-pass for x and the four weights)
// ---------------------------------------------------------------------------
__global__ __launch_bounds__(256) void cvt_kernel(const float* __restrict__ in,
                                                  bf16* __restrict__ out) {
  const size_t i = (size_t)blockIdx.x * 256 + threadIdx.x;   // 8 elements per thread
  const float4* p = reinterpret_cast<const float4*>(in) + i * 2;
  float4 a = p[0], b = p[1];
  union { bf16 h[8]; uint4 u; } r;
  r.h[0] = (bf16)a.x; r.h[1] = (bf16)a.y; r.h[2] = (bf16)a.z; r.h[3] = (bf16)a.w;
  r.h[4] = (bf16)b.x; r.h[5] = (bf16)b.y; r.h[6] = (bf16)b.z; r.h[7] = (bf16)b.w;
  reinterpret_cast<uint4*>(out)[i] = r.u;
}

// ---------------------------------------------------------------------------
// GEMM: out[m,n] = sum_k A[m,k] * W[n,k] + bias[n]   (A,W bf16; TDM-staged LDS)
// mode 0: epilogue RoPE+ReLU, store bf16 (B,H,S,D)   (q / k)
// mode 1: plain epilogue,     store bf16 (B,H,S,D)   (v)
// mode 2: bias epilogue,      store fp32 row-major   (output proj)
// ---------------------------------------------------------------------------
__global__ __launch_bounds__(256) void gemm_proj(
    const bf16* __restrict__ A, const bf16* __restrict__ W,
    const float* __restrict__ bias,
    const float* __restrict__ cosT, const float* __restrict__ sinT,
    bf16* __restrict__ out16, float* __restrict__ out32, int mode)
{
  __shared__ bf16 As[2][128 * LDA];
  __shared__ bf16 Ws[2][128 * LDA];

  const int tid  = threadIdx.x;
  const int m0   = blockIdx.x * 128;
  const int n0   = blockIdx.y * 128;
  const int lane = tid & 31, wid = tid >> 5;
  const int half = lane >> 4, r16 = lane & 15;
  const int wm = wid & 3, wn = wid >> 2;     // 4x2 wave grid: 32-row x 64-col per wave

  v8f acc[2][4];
  #pragma unroll
  for (int i = 0; i < 2; ++i)
    #pragma unroll
    for (int j = 0; j < 4; ++j) acc[i][j] = (v8f)(0.0f);

  constexpr int KT = C / 32;                 // 64 k-steps

  auto stage = [&](int kt, int buf) {
    const int k0 = kt * 32;
#if HAVE_TDM
    if (wid == 0) {                          // one TDM issue per tile per block
      tdm_load_tile(A + (size_t)m0 * C + k0,
                    (unsigned)(uintptr_t)(const void*)&As[buf][0], Mtot);
      tdm_load_tile(W + (size_t)n0 * C + k0,
                    (unsigned)(uintptr_t)(const void*)&Ws[buf][0], C);
    }
#else
    #pragma unroll
    for (int i = 0; i < 2; ++i) {
      int lin = tid + i * 256;
      int row = lin >> 2, c8 = (lin & 3) * 8;
      *reinterpret_cast<uint4*>(&As[buf][row * LDA + c8]) =
          *reinterpret_cast<const uint4*>(&A[(size_t)(m0 + row) * C + k0 + c8]);
      *reinterpret_cast<uint4*>(&Ws[buf][row * LDA + c8]) =
          *reinterpret_cast<const uint4*>(&W[(size_t)(n0 + row) * C + k0 + c8]);
    }
#endif
  };

  auto compute = [&](int buf) {
    const int aoff0 = half * 8, aoff1 = 16 + half * 8;   // A-frag K chunks per lane half
    const int boff  = half * 8;                           // B-frag K chunks per lane half
    v16bf a0 = ld_frag(&As[buf][(wm * 32 + 0 * 16 + r16) * LDA + aoff0],
                       &As[buf][(wm * 32 + 0 * 16 + r16) * LDA + aoff1]);
    v16bf a1 = ld_frag(&As[buf][(wm * 32 + 1 * 16 + r16) * LDA + aoff0],
                       &As[buf][(wm * 32 + 1 * 16 + r16) * LDA + aoff1]);
    #pragma unroll
    for (int nt = 0; nt < 4; ++nt) {
      const bf16* bp = &Ws[buf][(wn * 64 + nt * 16 + r16) * LDA + boff];
      v16bf b = ld_frag(bp, bp + 16);
      acc[0][nt] = WMMA_BF16(a0, b, acc[0][nt]);
      acc[1][nt] = WMMA_BF16(a1, b, acc[1][nt]);
    }
  };

  stage(0, 0);
  for (int kt = 0; kt < KT; ++kt) {
#if HAVE_TDM
    if (wid == 0) __builtin_amdgcn_s_wait_tensorcnt(0);  // DMA for tile kt done
#endif
    __syncthreads();                                      // publish tile kt to all waves
    if (kt + 1 < KT) stage(kt + 1, (kt + 1) & 1);
    compute(kt & 1);
  }

  // epilogue
  #pragma unroll
  for (int mt = 0; mt < 2; ++mt)
    #pragma unroll
    for (int nt = 0; nt < 4; ++nt)
      #pragma unroll
      for (int e = 0; e < 8; ++e) {
        const int m = m0 + wm * 32 + mt * 16 + e + half * 8;
        const int n = n0 + wn * 64 + nt * 16 + r16;
        float val = acc[mt][nt][e] + bias[n];
        if (mode < 2) {
          const int s = m & (S - 1), bb = m >> 12;
          const int dcol = n & (D - 1), h = n >> 7;
          float res = val;
          if (mode == 0) {
            const float cc = cosT[s * D + dcol];
            const float ss = sinT[s * D + dcol];
            const float partner = __shfl_xor(val, 1, 32);    // RoPE pair = adjacent lane
            const float rot = (dcol & 1) ? partner : -partner;
            res = fmaxf(val * cc + rot * ss, 0.0f);
          }
          out16[(((size_t)bb * H + h) * S + s) * D + dcol] = (bf16)res;
        } else {
          out32[(size_t)m * C + n] = val;
        }
      }
}

// ---------------------------------------------------------------------------
// vk[n,d] = sum_s v_pad[s,n] * k[s,d]   per (b,h); M=144 (129 padded), N=128, K=S
// (transposed staging -> manual copies; TDM cannot transpose)
// ---------------------------------------------------------------------------
__global__ __launch_bounds__(256) void vk_kernel(
    const bf16* __restrict__ kb, const bf16* __restrict__ vb, bf16* __restrict__ vkb)
{
  __shared__ bf16 Av[2][144 * LDA];
  __shared__ bf16 Bk[2][128 * LDA];

  const int bh = blockIdx.x;
  const bf16* kp = kb + (size_t)bh * S * D;
  const bf16* vp = vb + (size_t)bh * S * D;

  const int tid  = threadIdx.x;
  const int lane = tid & 31, wid = tid >> 5;     // wave wid owns d-cols [wid*16, +16)
  const int half = lane >> 4, r16 = lane & 15;

  v8f acc[9];
  #pragma unroll
  for (int i = 0; i < 9; ++i) acc[i] = (v8f)(0.0f);

  auto stage = [&](int st, int buf) {
    const int s0 = st * 32;
    #pragma unroll
    for (int i = 0; i < 16; ++i) {
      int lin = tid + 256 * i;
      int sc = lin >> 7, dd = lin & 127;
      Bk[buf][dd * LDA + sc] = kp[(size_t)(s0 + sc) * D + dd];  // transpose: [d][s]
      Av[buf][dd * LDA + sc] = vp[(size_t)(s0 + sc) * D + dd];  // transpose: [n][s]
    }
    if (tid < 32) {
      #pragma unroll
      for (int rr = 128; rr < 144; ++rr)
        Av[buf][rr * LDA + tid] = (rr == 128) ? (bf16)1.0f : (bf16)0.0f;  // ones pad row
    }
  };

  auto compute = [&](int buf) {
    const int aoff0 = half * 8, aoff1 = 16 + half * 8;
    const bf16* bp = &Bk[buf][(wid * 16 + r16) * LDA + half * 8];
    v16bf b = ld_frag(bp, bp + 16);
    #pragma unroll
    for (int mt = 0; mt < 9; ++mt) {
      v16bf a = ld_frag(&Av[buf][(mt * 16 + r16) * LDA + aoff0],
                        &Av[buf][(mt * 16 + r16) * LDA + aoff1]);
      acc[mt] = WMMA_BF16(a, b, acc[mt]);
    }
  };

  constexpr int ST = S / 32;   // 128
  stage(0, 0);
  for (int st = 0; st < ST; ++st) {
    __syncthreads();
    if (st + 1 < ST) stage(st + 1, (st + 1) & 1);
    compute(st & 1);
  }

  #pragma unroll
  for (int mt = 0; mt < 9; ++mt)
    #pragma unroll
    for (int e = 0; e < 8; ++e) {
      const int nrow = mt * 16 + e + half * 8;
      const int dcol = wid * 16 + r16;
      vkb[((size_t)bh * 144 + nrow) * D + dcol] = (bf16)acc[mt][e];
    }
}

// ---------------------------------------------------------------------------
// hs[s,n] = sum_d q[s,d]*vk[n,d]; attn[s,d] = hs[s,d]/(hs[s,128]+eps); K=128 only
// ---------------------------------------------------------------------------
__global__ __launch_bounds__(256) void attn_kernel(
    const bf16* __restrict__ qb, const bf16* __restrict__ vkb, bf16* __restrict__ attn)
{
  const int s0 = blockIdx.x * 128;
  const int bh = blockIdx.y;
  const int tid  = threadIdx.x;
  const int lane = tid & 31, wid = tid >> 5;     // wave owns rows [s0+wid*16, +16)
  const int half = lane >> 4, r16 = lane & 15;

  v8f acc[9];
  #pragma unroll
  for (int i = 0; i < 9; ++i) acc[i] = (v8f)(0.0f);

  const bf16* qbase  = qb  + (size_t)bh * S * D;
  const bf16* vkbase = vkb + (size_t)bh * 144 * D;

  #pragma unroll
  for (int ks = 0; ks < 4; ++ks) {
    const int k0 = ks * 32;
    const bf16* ap = qbase + (size_t)(s0 + wid * 16 + r16) * D + k0;
    v16bf a = ld_frag(ap + half * 8, ap + 16 + half * 8);
    #pragma unroll
    for (int nt = 0; nt < 9; ++nt) {
      const bf16* bp = vkbase + (size_t)(nt * 16 + r16) * D + k0 + half * 8;
      v16bf b = ld_frag(bp, bp + 16);
      acc[nt] = WMMA_BF16(a, b, acc[nt]);
    }
  }

  const int b = bh >> 4, h = bh & 15;
  #pragma unroll
  for (int e = 0; e < 8; ++e) {
    const float den = __shfl(acc[8][e], lane & 16, 32);   // n==128 column of this row
    const float inv = 1.0f / (den + 1e-15f);
    const int s = s0 + wid * 16 + e + half * 8;
    #pragma unroll
    for (int nt = 0; nt < 8; ++nt) {
      const float val = acc[nt][e] * inv;
      const int n = nt * 16 + r16;
      attn[((size_t)b * S + s) * C + h * D + n] = (bf16)val;
    }
  }
}

// ---------------------------------------------------------------------------
extern "C" void kernel_launch(void* const* d_in, const int* in_sizes, int n_in,
                              void* d_out, int out_size, void* d_ws, size_t ws_size,
                              hipStream_t stream) {
  const float* x    = (const float*)d_in[0];
  const float* cosT = (const float*)d_in[1];
  const float* sinT = (const float*)d_in[2];
  const float* Wq   = (const float*)d_in[3];
  const float* bq   = (const float*)d_in[4];
  const float* Wk   = (const float*)d_in[5];
  const float* bk   = (const float*)d_in[6];
  const float* Wv   = (const float*)d_in[7];
  const float* bv   = (const float*)d_in[8];
  const float* Wo   = (const float*)d_in[9];
  const float* bo   = (const float*)d_in[10];

  char* w = (char*)d_ws;
  const size_t big   = (size_t)Bc * H * S * D * sizeof(bf16);   // 33,554,432
  const size_t wsize = (size_t)C * C * sizeof(bf16);            //  8,388,608
  bf16* xb    = (bf16*)(w);                    // also reused as attn buffer later
  bf16* qb    = (bf16*)(w + big);
  bf16* kb    = (bf16*)(w + 2 * big);
  bf16* vb    = (bf16*)(w + 3 * big);
  bf16* wqb   = (bf16*)(w + 4 * big);
  bf16* wkb   = (bf16*)(w + 4 * big + wsize);
  bf16* wvb   = (bf16*)(w + 4 * big + 2 * wsize);
  bf16* wob   = (bf16*)(w + 4 * big + 3 * wsize);
  bf16* vkb   = (bf16*)(w + 4 * big + 4 * wsize);
  bf16* attnb = xb;                            // x is dead once q/k/v are built

  // one-time fp32 -> bf16 packs (x + 4 weights); 8 elems/thread
  cvt_kernel<<<(Mtot * C) / 2048, 256, 0, stream>>>(x,  xb);
  cvt_kernel<<<(C * C) / 2048,    256, 0, stream>>>(Wq, wqb);
  cvt_kernel<<<(C * C) / 2048,    256, 0, stream>>>(Wk, wkb);
  cvt_kernel<<<(C * C) / 2048,    256, 0, stream>>>(Wv, wvb);
  cvt_kernel<<<(C * C) / 2048,    256, 0, stream>>>(Wo, wob);

  dim3 gp(Mtot / 128, C / 128);   // 64 x 16
  gemm_proj<<<gp, 256, 0, stream>>>(xb, wqb, bq, cosT, sinT, qb, nullptr, 0);
  gemm_proj<<<gp, 256, 0, stream>>>(xb, wkb, bk, cosT, sinT, kb, nullptr, 0);
  gemm_proj<<<gp, 256, 0, stream>>>(xb, wvb, bv, nullptr, nullptr, vb, nullptr, 1);
  vk_kernel<<<Bc * H, 256, 0, stream>>>(kb, vb, vkb);
  attn_kernel<<<dim3(S / 128, Bc * H), 256, 0, stream>>>(qb, vkb, attnb);
  gemm_proj<<<gp, 256, 0, stream>>>(attnb, wob, bo, nullptr, nullptr, nullptr,
                                    (float*)d_out, 2);
}
